// HyperDecoderBlock_66236985639771
// MI455X (gfx1250) — matrix operand order repaired and checked by reading, not compiled, not run on hardware
//
#include <hip/hip_runtime.h>
#include <math.h>

// ---------------------------------------------------------------------------
// CDNA5 WMMA (wave32): D(16x16 f32) = A(16x32 bf16) x B(32x16 bf16) + C
// ---------------------------------------------------------------------------
typedef __attribute__((ext_vector_type(16))) __bf16  v16bf;
typedef __attribute__((ext_vector_type(8)))  float   v8f;
typedef __attribute__((ext_vector_type(8)))  unsigned int v8u;

__device__ inline unsigned short f2bf(float f) {   // RNE float -> bf16
  unsigned int u = __builtin_bit_cast(unsigned int, f);
  return (unsigned short)((u + 0x7FFFu + ((u >> 16) & 1u)) >> 16);
}
__device__ inline unsigned pack_bf2(float a, float b) {  // {lo=a, hi=b}
  unsigned ua = __builtin_bit_cast(unsigned, a);
  unsigned ub = __builtin_bit_cast(unsigned, b);
  ua += 0x7FFFu + ((ua >> 16) & 1u);
  ub += 0x7FFFu + ((ub >> 16) & 1u);
  return (ua >> 16) | (ub & 0xFFFF0000u);
}
__device__ inline v8f vzero() {
  v8f z;
#pragma unroll
  for (int i = 0; i < 8; ++i) z[i] = 0.0f;
  return z;
}
__device__ inline v8f wmma_bf16(v8u a, v8u b, v8f c) {
  return __builtin_amdgcn_wmma_f32_16x16x32_bf16(
      false, __builtin_bit_cast(v16bf, a),
      false, __builtin_bit_cast(v16bf, b),
      (short)0, c, false, false);
}
__device__ inline v8u mk8(uint4 a, uint4 b) {
  v8u r;
  r[0] = a.x; r[1] = a.y; r[2] = a.z; r[3] = a.w;
  r[4] = b.x; r[5] = b.y; r[6] = b.z; r[7] = b.w;
  return r;
}

// ----- CDNA5 async global->LDS copy (ASYNCcnt-tracked, no VGPR round-trip) --
__device__ inline unsigned lds_off(const void* p) {
  // Generic LDS address = {shared aperture, offset[31:0]} (ISA 10.2)
  return (unsigned)(size_t)p;
}
__device__ inline void async_copy_b128(const void* gsrc, unsigned dst_lds) {
  asm volatile("global_load_async_to_lds_b128 %0, %1, off"
               :: "v"(dst_lds), "v"(gsrc) : "memory");
}
__device__ inline void wait_async() {
  asm volatile("s_wait_asynccnt 0" ::: "memory");
}

#define BB 2
#define SS 2048
#define DD 2048
#define HH 16
#define DHH 128
#define DFFN 8192
#define TT (BB * SS)      // 4096 tokens
#define QAP 80            // padded Lorentz feature pairs (160 feats / 2)
#define EPS_F 1e-6f

// ---------------------------------------------------------------------------
// pack_rows: contiguous fp32 pairs -> bf16-pair uints (A-side activations)
// ---------------------------------------------------------------------------
__global__ __launch_bounds__(256) void hyp_pack_rows_kernel(
    const float* __restrict__ in, unsigned* __restrict__ out) {
  const size_t i = (size_t)blockIdx.x * 256 + threadIdx.x;  // one uint4 each
  const float4 f0 = *(const float4*)(in + i * 8);
  const float4 f1 = *(const float4*)(in + i * 8 + 4);
  uint4 o;
  o.x = pack_bf2(f0.x, f0.y);
  o.y = pack_bf2(f0.z, f0.w);
  o.z = pack_bf2(f1.x, f1.y);
  o.w = pack_bf2(f1.z, f1.w);
  *(uint4*)(out + i * 4) = o;
}

// ---------------------------------------------------------------------------
// pack_wblocked: W[K][N] fp32 -> WT[K/32][N][16] bf16-pair uints.  One-time.
// ---------------------------------------------------------------------------
__global__ __launch_bounds__(256) void hyp_pack_wblocked_kernel(
    const float* __restrict__ W, unsigned* __restrict__ WT, int nshift) {
  const size_t gid = (size_t)blockIdx.x * 256 + threadIdx.x;  // one uint each
  const int N = 1 << nshift;
  const int p = (int)(gid & 15);
  const int c = (int)((gid >> 4) & (N - 1));
  const size_t kb = gid >> (4 + nshift);
  const size_t r = kb * 32 + 2 * p;
  WT[gid] = pack_bf2(W[r * N + c], W[(r + 1) * N + c]);
}

// ---------------------------------------------------------------------------
// pack_vblocked: V[T][D] fp32 -> VPB[b][S/32][H][128][16] bf16-pair uints.
// ---------------------------------------------------------------------------
__global__ __launch_bounds__(256) void hyp_pack_vblocked_kernel(
    const float* __restrict__ vb, unsigned* __restrict__ VPB) {
  const size_t gid = (size_t)blockIdx.x * 256 + threadIdx.x;  // one uint each
  const int p = (int)(gid & 15);
  const int c = (int)((gid >> 4) & 127);
  const int h = (int)((gid >> 11) & 15);
  const int kb = (int)((gid >> 15) & 63);
  const int b = (int)(gid >> 21);
  const size_t row = (size_t)b * SS + kb * 32 + 2 * p;
  VPB[gid] = pack_bf2(vb[row * DD + h * DHH + c],
                      vb[(row + 1) * DD + h * DHH + c]);
}

// ---------------------------------------------------------------------------
// log-map + Lorentz RMSNorm: u fp32, vn bf16-pair packed.  One block/token.
// ---------------------------------------------------------------------------
__global__ __launch_bounds__(256) void hyp_lognorm_kernel(
    const float* __restrict__ x, const float* __restrict__ scale,
    float* __restrict__ u, unsigned* __restrict__ vnP) {
  __shared__ float red[256];
  const int t = blockIdx.x;
  const int tid = threadIdx.x;
  const int i0 = tid * 8;
  const float* xp = x + (size_t)t * (DD + 1) + 1;  // unaligned base: scalar ld
  float xi[8];
  float s = 0.0f;
#pragma unroll
  for (int i = 0; i < 8; ++i) {
    xi[i] = xp[i0 + i];
    s += xi[i] * xi[i];
  }
  red[tid] = s;
  __syncthreads();
  for (int st = 128; st >= 1; st >>= 1) {
    if (tid < st) red[tid] += red[tid + st];
    __syncthreads();
  }
  const float n2 = red[0];
  const float n = sqrtf(n2);
  const float f = (n < 1e-6f) ? (1.0f - n2 * (1.0f / 6.0f))
                              : (asinhf(n) / fmaxf(n, 1e-12f));
  const float rinv = rsqrtf(f * f * n2 * (1.0f / (float)DD) + EPS_F);
  float uu[8], vv[8];
#pragma unroll
  for (int i = 0; i < 8; ++i) {
    uu[i] = f * xi[i];
    vv[i] = scale[i0 + i] * uu[i] * rinv;
  }
  float4* up = (float4*)(u + (size_t)t * DD + i0);
  up[0] = make_float4(uu[0], uu[1], uu[2], uu[3]);
  up[1] = make_float4(uu[4], uu[5], uu[6], uu[7]);
  uint4 o;
  o.x = pack_bf2(vv[0], vv[1]);
  o.y = pack_bf2(vv[2], vv[3]);
  o.z = pack_bf2(vv[4], vv[5]);
  o.w = pack_bf2(vv[6], vv[7]);
  *(uint4*)(vnP + (size_t)t * (DD / 2) + tid * 4) = o;
}

// ---------------------------------------------------------------------------
// GEMM: C[M,N] = act( A @ W ).  A: bf16 pairs [M][K/2]; W: blocked
// WT[K/32][N][16].  128x128 tile, 8 waves, double-buffered LDS with CDNA5
// async global->LDS staging.  Fragments: 2x ds_load_b128 each, feed WMMA
// operand quads directly.
// ---------------------------------------------------------------------------
__global__ __launch_bounds__(256) void hyp_gemm_kernel(
    const unsigned* __restrict__ AP, const unsigned* __restrict__ WT,
    float* __restrict__ C, int M, int N, int K, int gelu) {
  __shared__ __align__(16) unsigned AsU[2][128][20];  // [row][16 pairs]+pad
  __shared__ __align__(16) unsigned BsT[2][128][20];  // [col][16 pairs]+pad
  const int Kp = K >> 1;
  const int nkb = Kp >> 4;          // 32-element K blocks
  const int tid = threadIdx.x;
  const int lane = tid & 31;
  const int w = tid >> 5;
  const int wm = w >> 1;            // 0..3
  const int wn = w & 1;             // 0..1
  const int half = lane >> 4;
  const int l16 = lane & 15;
  const int N0 = blockIdx.x * 128;
  const int M0 = blockIdx.y * 128;

  auto loadA = [&](int buf, int kb) {
#pragma unroll
    for (int i = 0; i < 2; ++i) {
      const int g = tid + i * 256;
      const int r = g >> 2;
      const int c4 = (g & 3) << 2;
      async_copy_b128(AP + (size_t)(M0 + r) * Kp + kb * 16 + c4,
                      lds_off(&AsU[buf][r][c4]));
    }
  };
  auto loadB = [&](int buf, int kb) {
#pragma unroll
    for (int i = 0; i < 2; ++i) {
      const int g = tid + i * 256;
      const int c = g >> 2;
      const int p4 = (g & 3) << 2;
      async_copy_b128(WT + ((size_t)kb * N + N0 + c) * 16 + p4,
                      lds_off(&BsT[buf][c][p4]));
    }
  };

  v8f acc[2][4];
#pragma unroll
  for (int mf = 0; mf < 2; ++mf)
#pragma unroll
    for (int nf = 0; nf < 4; ++nf) acc[mf][nf] = vzero();

  loadA(0, 0);
  loadB(0, 0);
  for (int kb = 0; kb < nkb; ++kb) {
    wait_async();                   // our async copies for tile kb complete
    __syncthreads();                // everyone's complete
    if (kb + 1 < nkb) {             // start async staging of next tile
      loadA((kb + 1) & 1, kb + 1);
      loadB((kb + 1) & 1, kb + 1);
    }
    const int cur = kb & 1;

    v8u afr[2];
#pragma unroll
    for (int mf = 0; mf < 2; ++mf) {
      const int row = wm * 32 + mf * 16 + l16;
      const uint4 a0 = *(const uint4*)&AsU[cur][row][half * 4];
      const uint4 a1 = *(const uint4*)&AsU[cur][row][8 + half * 4];
      afr[mf] = mk8(a0, a1);
    }
#pragma unroll
    for (int nf = 0; nf < 4; ++nf) {
      const int col = wn * 64 + nf * 16 + l16;
      const uint4 b0 = *(const uint4*)&BsT[cur][col][half * 4];
      const uint4 b1 = *(const uint4*)&BsT[cur][col][8 + half * 4];
      const v8u bfr = mk8(b0, b1);
      acc[0][nf] = wmma_bf16(afr[0], bfr, acc[0][nf]);
      acc[1][nf] = wmma_bf16(afr[1], bfr, acc[1][nf]);
    }
  }

#pragma unroll
  for (int mf = 0; mf < 2; ++mf)
#pragma unroll
    for (int nf = 0; nf < 4; ++nf) {
      const int col = N0 + wn * 64 + nf * 16 + l16;
#pragma unroll
      for (int v = 0; v < 8; ++v) {
        const int row = M0 + wm * 32 + mf * 16 + v + (half << 3);
        float val = acc[mf][nf][v];
        if (gelu) val = 0.5f * val * (1.0f + erff(val * 0.70710678118654752f));
        C[(size_t)row * N + col] = val;
      }
    }
}

// ---------------------------------------------------------------------------
// Per-head exp-map of q,k -> bf16-pair packed Lorentz vectors, fragment-ready.
// Feature layout: [0..127]=xi, [128]=x0 term (negated for q), rest zero.
// ---------------------------------------------------------------------------
__global__ __launch_bounds__(256) void hyp_expmapqk_kernel(
    const float* __restrict__ q, const float* __restrict__ k,
    const float* __restrict__ headK, unsigned* __restrict__ qaP,
    unsigned* __restrict__ kaP) {
  const int lane = threadIdx.x & 31;
  const int w = threadIdx.x >> 5;
  const int u = blockIdx.x * 8 + w;   // 0 .. T*H-1
  const int t = u >> 4;
  const int h = u & 15;
  const float Kh = headK[h];
  const float sK = sqrtf(-Kh);

  const float4 qv = *(const float4*)(q + (size_t)t * DD + h * DHH + lane * 4);
  const float4 kv = *(const float4*)(k + (size_t)t * DD + h * DHH + lane * 4);
  float qn = qv.x * qv.x + qv.y * qv.y + qv.z * qv.z + qv.w * qv.w;
  float kn = kv.x * kv.x + kv.y * kv.y + kv.z * kv.z + kv.w * kv.w;
#pragma unroll
  for (int off = 1; off <= 16; off <<= 1) {
    qn += __shfl_xor(qn, off, 32);
    kn += __shfl_xor(kn, off, 32);
  }
  const float aq = sK * sqrtf(qn);
  const float ak = sK * sqrtf(kn);
  const float sq = (aq < 1e-6f) ? (1.0f + aq * aq * (1.0f / 6.0f))
                                : (sinhf(aq) / fmaxf(aq, 1e-12f));
  const float sk = (ak < 1e-6f) ? (1.0f + ak * ak * (1.0f / 6.0f))
                                : (sinhf(ak) / fmaxf(ak, 1e-12f));
  const float x0q = sqrtf(-1.0f / Kh + sq * sq * qn);
  const float x0k = sqrtf(-1.0f / Kh + sk * sk * kn);

  const int b = t / SS;
  const int sidx = t & (SS - 1);
  const size_t base = ((size_t)(b * HH + h) * SS + sidx) * QAP;
  unsigned* qd = qaP + base;
  unsigned* kd = kaP + base;
  uint2 qo, ko;
  qo.x = pack_bf2(sq * qv.x, sq * qv.y);
  qo.y = pack_bf2(sq * qv.z, sq * qv.w);
  ko.x = pack_bf2(sk * kv.x, sk * kv.y);
  ko.y = pack_bf2(sk * kv.z, sk * kv.w);
  *(uint2*)(qd + 2 * lane) = qo;
  *(uint2*)(kd + 2 * lane) = ko;
  if (lane == 0) {
    qd[64] = pack_bf2(-x0q, 0.0f);
    kd[64] = pack_bf2(x0k, 0.0f);
  } else if (lane < 16) {           // zero pad pairs 65..79
    qd[64 + lane] = 0u;
    kd[64 + lane] = 0u;
  }
}

// ---------------------------------------------------------------------------
// Causal flash attention.  One wave per 16-row q-tile; qa/ka fragments via
// raw global b128; V staged to per-wave LDS region with async copies.
// No __syncthreads inside loop (waves have different causal trip counts).
// ---------------------------------------------------------------------------
__global__ __launch_bounds__(128) void hyp_attn_kernel(
    const unsigned* __restrict__ qaP, const unsigned* __restrict__ kaP,
    const unsigned* __restrict__ VPB, const float* __restrict__ headK,
    float* __restrict__ ctx) {
  __shared__ __align__(16) unsigned VtT[4][128][20];  // [col][16 pairs]+pad
  __shared__ __align__(16) unsigned PtU[4][16][20];   // [row][16 col-pairs]
  const int tid = threadIdx.x;
  const int lane = tid & 31;
  const int w = tid >> 5;
  const int half = lane >> 4;
  const int l16 = lane & 15;
  const int bh = blockIdx.y;
  const int b = bh >> 4;
  const int h = bh & 15;
  const int qbase = (blockIdx.x * 4 + w) * 16;

  const float Kh = headK[h];
  const float inv_sqrt_dh = 0.08838834764831845f;  // 1/sqrt(128)
  const float c1 = 2.0f * inv_sqrt_dh;
  const float c2 = (-2.0f / Kh) * inv_sqrt_dh;
  const unsigned* qah = qaP + (size_t)bh * SS * QAP;
  const unsigned* kah = kaP + (size_t)bh * SS * QAP;

  // Preload 5 A-fragments of qa (16 x 160 feats): 2 global b128 each
  v8u qfr[5];
  {
    const size_t rb = (size_t)(qbase + l16) * QAP;
#pragma unroll
    for (int f = 0; f < 5; ++f) {
      const uint4 a0 = *(const uint4*)(qah + rb + f * 16 + half * 4);
      const uint4 a1 = *(const uint4*)(qah + rb + f * 16 + 8 + half * 4);
      qfr[f] = mk8(a0, a1);
    }
  }

  v8f out[8];
  float m[8], lsum[8];
#pragma unroll
  for (int dt = 0; dt < 8; ++dt) out[dt] = vzero();
#pragma unroll
  for (int v = 0; v < 8; ++v) { m[v] = -3.0e38f; lsum[v] = 0.0f; }

  const int nk = (qbase + 16 + 31) >> 5;   // 32-key chunks (causal)
  for (int c = 0; c < nk; ++c) {
    const int kb = c * 32;

    // Async-stage V chunk (per-wave LDS region; waited before P@V below)
    const size_t vbase =
        (((size_t)(b * (SS / 32) + (kb >> 5)) * HH + h) * DHH) * 16;
#pragma unroll
    for (int i = 0; i < 16; ++i) {
      const int g = lane + i * 32;
      const int cc = g >> 2;
      const int p4 = (g & 3) << 2;
      async_copy_b128(VPB + vbase + cc * 16 + p4, lds_off(&VtT[w][cc][p4]));
    }

    // Scores: 2 x 16x16 tiles, 5 feature chunks; ka frags via global b128
    v8f s[2];
#pragma unroll
    for (int nt = 0; nt < 2; ++nt) {
      v8f a = vzero();
      const size_t cb = (size_t)(kb + nt * 16 + l16) * QAP;
#pragma unroll
      for (int f = 0; f < 5; ++f) {
        const uint4 b0 = *(const uint4*)(kah + cb + f * 16 + half * 4);
        const uint4 b1 = *(const uint4*)(kah + cb + f * 16 + 8 + half * 4);
        a = wmma_bf16(qfr[f], mk8(b0, b1), a);
      }
      s[nt] = a;
    }

    // Scale + causal mask + online softmax
    float mv[8];
#pragma unroll
    for (int v = 0; v < 8; ++v) {
      const int qrow = qbase + v + (half << 3);
      float s0 = c1 * s[0][v] + c2;
      float s1 = c1 * s[1][v] + c2;
      if (kb + l16 > qrow) s0 = -1.0e30f;
      if (kb + 16 + l16 > qrow) s1 = -1.0e30f;
      s[0][v] = s0;
      s[1][v] = s1;
      mv[v] = fmaxf(s0, s1);
    }
#pragma unroll
    for (int off = 1; off <= 8; off <<= 1)
#pragma unroll
      for (int v = 0; v < 8; ++v)
        mv[v] = fmaxf(mv[v], __shfl_xor(mv[v], off, 32));

    float alpha[8], ps[8];
#pragma unroll
    for (int v = 0; v < 8; ++v) {
      const float mn = fmaxf(m[v], mv[v]);
      alpha[v] = __expf(m[v] - mn);
      m[v] = mn;
      const float p0 = __expf(s[0][v] - mn);
      const float p1 = __expf(s[1][v] - mn);
      ps[v] = p0 + p1;
      unsigned short* prow =
          (unsigned short*)&PtU[w][v + (half << 3)][0];  // ushort idx == col
      prow[l16] = f2bf(p0);
      prow[16 + l16] = f2bf(p1);
    }
#pragma unroll
    for (int off = 1; off <= 8; off <<= 1)
#pragma unroll
      for (int v = 0; v < 8; ++v) ps[v] += __shfl_xor(ps[v], off, 32);
#pragma unroll
    for (int v = 0; v < 8; ++v) lsum[v] = lsum[v] * alpha[v] + ps[v];
#pragma unroll
    for (int dt = 0; dt < 8; ++dt)
#pragma unroll
      for (int v = 0; v < 8; ++v) out[dt][v] *= alpha[v];

    // P as A-fragment: 2 ds_load_b128
    const uint4 p0 = *(const uint4*)&PtU[w][l16][half * 4];
    const uint4 p1 = *(const uint4*)&PtU[w][l16][8 + half * 4];
    const v8u pfr = mk8(p0, p1);

    wait_async();                   // V tile resident in LDS

    // out += P @ V : 8 WMMAs, V frags = 2x ds_load_b128 each
#pragma unroll
    for (int dt = 0; dt < 8; ++dt) {
      const int col = dt * 16 + l16;
      const uint4 b0 = *(const uint4*)&VtT[w][col][half * 4];
      const uint4 b1 = *(const uint4*)&VtT[w][col][8 + half * 4];
      out[dt] = wmma_bf16(pfr, mk8(b0, b1), out[dt]);
    }
  }

  // Normalize + write ctx token-major (B*S, H*DH)
#pragma unroll
  for (int v = 0; v < 8; ++v) {
    const float inv = 1.0f / lsum[v];
    const int row = qbase + v + (half << 3);
    float* dst = ctx + ((size_t)(b * SS + row)) * DD + h * DHH;
#pragma unroll
    for (int dt = 0; dt < 8; ++dt) dst[dt * 16 + l16] = out[dt][v] * inv;
  }
}

// ---------------------------------------------------------------------------
// w = u1 + attn_out; vn = scale * w / rms(w)  (fp32 w + packed vn)
// ---------------------------------------------------------------------------
__global__ __launch_bounds__(256) void hyp_addnorm_kernel(
    const float* __restrict__ u1, const float* __restrict__ ao,
    const float* __restrict__ scale, float* __restrict__ wout,
    unsigned* __restrict__ vnP) {
  __shared__ float red[256];
  const int t = blockIdx.x;
  const int tid = threadIdx.x;
  const int i0 = tid * 8;
  float wv[8];
  float s = 0.0f;
#pragma unroll
  for (int i = 0; i < 8; ++i) {
    const int idx = i0 + i;
    wv[i] = u1[(size_t)t * DD + idx] + ao[(size_t)t * DD + idx];
    s += wv[i] * wv[i];
  }
  red[tid] = s;
  __syncthreads();
  for (int st = 128; st >= 1; st >>= 1) {
    if (tid < st) red[tid] += red[tid + st];
    __syncthreads();
  }
  const float rinv = rsqrtf(red[0] * (1.0f / (float)DD) + EPS_F);
  float vv[8];
#pragma unroll
  for (int i = 0; i < 8; ++i) vv[i] = scale[i0 + i] * wv[i] * rinv;
  float4* wp = (float4*)(wout + (size_t)t * DD + i0);
  wp[0] = make_float4(wv[0], wv[1], wv[2], wv[3]);
  wp[1] = make_float4(wv[4], wv[5], wv[6], wv[7]);
  uint4 o;
  o.x = pack_bf2(vv[0], vv[1]);
  o.y = pack_bf2(vv[2], vv[3]);
  o.z = pack_bf2(vv[4], vv[5]);
  o.w = pack_bf2(vv[6], vv[7]);
  *(uint4*)(vnP + (size_t)t * (DD / 2) + tid * 4) = o;
}

// ---------------------------------------------------------------------------
// out = exp_map(w + ffn_pre, K=-1) -> (D+1)-dim Lorentz point
// ---------------------------------------------------------------------------
__global__ __launch_bounds__(256) void hyp_final_kernel(
    const float* __restrict__ wbuf, const float* __restrict__ f2,
    float* __restrict__ out) {
  __shared__ float red[256];
  const int t = blockIdx.x;
  const int tid = threadIdx.x;
  const int i0 = tid * 8;
  float y[8];
  float s = 0.0f;
#pragma unroll
  for (int i = 0; i < 8; ++i) {
    const int idx = i0 + i;
    y[i] = wbuf[(size_t)t * DD + idx] + f2[(size_t)t * DD + idx];
    s += y[i] * y[i];
  }
  red[tid] = s;
  __syncthreads();
  for (int st = 128; st >= 1; st >>= 1) {
    if (tid < st) red[tid] += red[tid + st];
    __syncthreads();
  }
  const float n2 = red[0];
  const float a = sqrtf(n2);
  const float sc = (a < 1e-6f) ? (1.0f + a * a * (1.0f / 6.0f))
                               : (sinhf(a) / fmaxf(a, 1e-12f));
  const float x0 = sqrtf(1.0f + sc * sc * n2);
  float* op = out + (size_t)t * (DD + 1);
  if (tid == 0) op[0] = x0;
#pragma unroll
  for (int i = 0; i < 8; ++i) op[1 + i0 + i] = sc * y[i];
}

// ---------------------------------------------------------------------------
// Host launcher
// ---------------------------------------------------------------------------
extern "C" void kernel_launch(void* const* d_in, const int* in_sizes, int n_in,
                              void* d_out, int out_size, void* d_ws,
                              size_t ws_size, hipStream_t stream) {
  const float* x      = (const float*)d_in[0];
  // d_in[1] = causal mask (computed analytically)
  const float* scale1 = (const float*)d_in[2];
  const float* scale2 = (const float*)d_in[3];
  const float* Wq     = (const float*)d_in[4];
  const float* Wk     = (const float*)d_in[5];
  const float* Wv     = (const float*)d_in[6];
  const float* Wo     = (const float*)d_in[7];
  const float* headK  = (const float*)d_in[8];
  const float* W1     = (const float*)d_in[9];
  const float* W2     = (const float*)d_in[10];
  float* outp = (float*)d_out;

  char* ws = (char*)d_ws;
  size_t off = 0;
  auto take = [&](size_t bytes) {
    void* p = ws + off;
    off += (bytes + 255) & ~(size_t)255;
    return p;
  };
  const size_t TD  = (size_t)TT * DD * sizeof(float);            // 32 MiB
  const size_t TDp = TD / 2;
  const size_t QAb = (size_t)BB * HH * SS * QAP * sizeof(unsigned);

  float*    u1   = (float*)take(TD);       // u1, later ffn_pre (f2)
  unsigned* vnP  = (unsigned*)take(TDp);   // vn1 then vn2 (packed)
  float*    qb   = (float*)take(TD);       // q, later ctx
  float*    kb   = (float*)take(TD);       // k, later attn_out
  float*    vb   = (float*)take(TD);       // v, later residual w
  unsigned* qaP  = (unsigned*)take(QAb);   // also reused for ctxP
  unsigned* kaP  = (unsigned*)take(QAb);
  unsigned* VPB  = (unsigned*)take(TDp);
  float*    f1   = (float*)take((size_t)TT * DFFN * sizeof(float));
  unsigned* f1P  = (unsigned*)take((size_t)TT * DFFN / 2 * sizeof(unsigned));
  unsigned* WqT  = (unsigned*)take((size_t)(DD / 2) * DD * sizeof(unsigned));
  unsigned* WkT  = (unsigned*)take((size_t)(DD / 2) * DD * sizeof(unsigned));
  unsigned* WvT  = (unsigned*)take((size_t)(DD / 2) * DD * sizeof(unsigned));
  unsigned* WoT  = (unsigned*)take((size_t)(DD / 2) * DD * sizeof(unsigned));
  unsigned* W1T  = (unsigned*)take((size_t)(DD / 2) * DFFN * sizeof(unsigned));
  unsigned* W2T  = (unsigned*)take((size_t)(DFFN / 2) * DD * sizeof(unsigned));
  unsigned* ctxP = qaP;                    // attention done before ctx pack

  const int gWdd = (int)((size_t)DD * DD / 2 / 256);       // 8192 blocks
  const int gWff = (int)((size_t)DD * DFFN / 2 / 256);     // 32768 blocks
  const dim3 gD(DD / 128, TT / 128);                       // (16, 32)
  const dim3 gF(DFFN / 128, TT / 128);                     // (64, 32)

  // One-time blocked bf16 packing of weights: WT[K/32][N][16]
  hyp_pack_wblocked_kernel<<<gWdd, 256, 0, stream>>>(Wq, WqT, 11);
  hyp_pack_wblocked_kernel<<<gWdd, 256, 0, stream>>>(Wk, WkT, 11);
  hyp_pack_wblocked_kernel<<<gWdd, 256, 0, stream>>>(Wv, WvT, 11);
  hyp_pack_wblocked_kernel<<<gWdd, 256, 0, stream>>>(Wo, WoT, 11);
  hyp_pack_wblocked_kernel<<<gWff, 256, 0, stream>>>(W1, W1T, 13);
  hyp_pack_wblocked_kernel<<<gWff, 256, 0, stream>>>(W2, W2T, 11);

  hyp_lognorm_kernel<<<TT, 256, 0, stream>>>(x, scale1, u1, vnP);
  hyp_gemm_kernel<<<gD, 256, 0, stream>>>(vnP, WqT, qb, TT, DD, DD, 0);
  hyp_gemm_kernel<<<gD, 256, 0, stream>>>(vnP, WkT, kb, TT, DD, DD, 0);
  hyp_gemm_kernel<<<gD, 256, 0, stream>>>(vnP, WvT, vb, TT, DD, DD, 0);
  hyp_expmapqk_kernel<<<(TT * HH) / 8, 256, 0, stream>>>(qb, kb, headK, qaP, kaP);
  hyp_pack_vblocked_kernel<<<(int)((size_t)TT * DD / 2 / 256), 256, 0, stream>>>(
      vb, VPB);
  hyp_attn_kernel<<<dim3((SS / 16) / 4, BB * HH), 128, 0, stream>>>(
      qaP, kaP, VPB, headK, qb /* ctx */);
  hyp_pack_rows_kernel<<<TT * DD / 8 / 256, 256, 0, stream>>>(qb, ctxP);
  hyp_gemm_kernel<<<gD, 256, 0, stream>>>(ctxP, WoT, kb /* attn_out */,
                                          TT, DD, DD, 0);
  hyp_addnorm_kernel<<<TT, 256, 0, stream>>>(u1, kb, scale2, vb /* w */, vnP);
  hyp_gemm_kernel<<<gF, 256, 0, stream>>>(vnP, W1T, f1, TT, DFFN, DD, 1);
  hyp_pack_rows_kernel<<<TT * DFFN / 8 / 256, 256, 0, stream>>>(f1, f1P);
  hyp_gemm_kernel<<<gD, 256, 0, stream>>>(f1P, W2T, u1 /* ffn_pre */,
                                          TT, DD, DFFN, 0);
  hyp_final_kernel<<<TT, 256, 0, stream>>>(vb, u1, outp);
}
